// MultiHeadAttention_35605278884376
// MI455X (gfx1250) — matrix operand order repaired and checked by reading, not compiled, not run on hardware
//
#include <hip/hip_runtime.h>
#include <hip/hip_bf16.h>

// ---------------------------------------------------------------------------
// MHA for MI455X (gfx1250): bf16 WMMA GEMMs + flash attention + TDM staging.
// B=2, S=2048, D=1024, H=16, DK=64.
// ---------------------------------------------------------------------------

#define B_  2
#define S_  2048
#define D_  1024
#define H_  16
#define DK_ 64
#define MROWS_ (B_ * S_)   // 4096

typedef __attribute__((ext_vector_type(16))) __bf16 v16bf;
typedef __attribute__((ext_vector_type(8)))  float  v8f;

__device__ __forceinline__ __bf16 to_bf16(float f)  { return (__bf16)f; }  // native v_cvt
__device__ __forceinline__ __bf16 to_bf16(__bf16 h) { return h; }

__device__ __forceinline__ v8f vzero8() {
  v8f z = {0.f, 0.f, 0.f, 0.f, 0.f, 0.f, 0.f, 0.f};
  return z;
}

__device__ __forceinline__ v8f wmma_bf16(v16bf a, v16bf b, v8f c) {
  return __builtin_amdgcn_wmma_f32_16x16x32_bf16(
      /*neg_a=*/false, a, /*neg_b=*/false, b,
      /*c_mod=*/(short)0, c, /*reuse_a=*/false, /*reuse_b=*/false);
}

// A fragment (16x32, M x K): element e -> k = (e&7) + 8*half + 16*(e>>3)
__device__ __forceinline__ v16bf load_frag_a(const __bf16* row, int half) {
  v16bf f;
#pragma unroll
  for (int e = 0; e < 16; ++e)
    f[e] = row[(e & 7) + 8 * half + 16 * (e >> 3)];
  return f;
}
// B fragment (32x16, K x N) from [n][k] row-major: element e -> k = (e&7) + 8*half + 16*(e>>3)
__device__ __forceinline__ v16bf load_frag_b(const __bf16* row, int half) {
  v16bf f;
#pragma unroll
  for (int e = 0; e < 16; ++e)
    f[e] = row[(e & 7) + 8 * half + 16 * (e >> 3)];
  return f;
}

// ---------------------------------------------------------------------------
// Tensor Data Mover: 2D bf16 tile (tile_d0 x tile_d1) -> LDS, with row padding
// so LDS row stride = tile_d0 + 8 bf16.  D# layout per CDNA5 ISA ch.8.
// Toolchain probe: this clang exposes the 6-arg builtin
//   (u32x4 g0, i32x8 g1, i32x4 g2, i32x4 g3, i32x8 g4, i32 cpol).
// ---------------------------------------------------------------------------
#if defined(__has_builtin)
#if __has_builtin(__builtin_amdgcn_tensor_load_to_lds)
#define HAVE_TDM 1
#endif
#endif

#ifdef HAVE_TDM
typedef unsigned int u32x4 __attribute__((ext_vector_type(4)));
typedef int          i32x8 __attribute__((ext_vector_type(8)));
typedef int          i32x4 __attribute__((ext_vector_type(4)));

__device__ __forceinline__ void tdm_load_2d_bf16(
    unsigned lds_addr, const void* gaddr,
    unsigned tensor_d0, unsigned tensor_d1,
    unsigned tile_d0, unsigned tile_d1, unsigned stride0_elems) {
  const unsigned long long ga = (unsigned long long)(uintptr_t)gaddr;
  u32x4 g0;
  g0[0] = 1u;                                        // count=1, user descriptor
  g0[1] = lds_addr;                                  // LDS byte address
  g0[2] = (unsigned)(ga & 0xFFFFFFFFu);              // global_addr[31:0]
  g0[3] = (unsigned)((ga >> 32) & 0x01FFFFFFu) | (2u << 30);  // [56:32] | type=2
  i32x8 g1;
  // data_size=1 (2B) | pad_enable | pad_interval=4 (32 DW = 1 row of 64 bf16)
  // | pad_amount=3 (4 DW = 8 bf16)  -> LDS row stride 72 bf16
  g1[0] = (int)((1u << 16) | (1u << 20) | (4u << 22) | (3u << 25));
  g1[1] = (int)((tensor_d0 & 0xFFFFu) << 16);                        // dim0 lo16
  g1[2] = (int)(((tensor_d0 >> 16) & 0xFFFFu) | ((tensor_d1 & 0xFFFFu) << 16));
  g1[3] = (int)(((tensor_d1 >> 16) & 0xFFFFu) | ((tile_d0 & 0xFFFFu) << 16));
  g1[4] = (int)(tile_d1 & 0xFFFFu);                                  // tile_dim2=0
  g1[5] = (int)stride0_elems;                                        // dim0 stride
  g1[6] = 0;
  g1[7] = 0;
  const i32x4 gz4 = {0, 0, 0, 0};
  const i32x8 gz8 = {0, 0, 0, 0, 0, 0, 0, 0};
  __builtin_amdgcn_tensor_load_to_lds(g0, g1, gz4, gz4, gz8, 0);
}
#endif

// ---------------------------------------------------------------------------
// GEMM: out[m][n] = sum_k A[m][k] * W[n][k] + bias[n]
// Block tile 128x128, 8 waves (4x2), each wave 32x64 (2x4 WMMA frags).
// OUTMODE 0: store bf16 head-split [B,H,S,DK].  OUTMODE 1: store f32 [M,N].
// ---------------------------------------------------------------------------
template <typename AT, int OUTMODE>
__global__ __launch_bounds__(256) void gemm_bf16_wmma(
    const AT* __restrict__ A, const float* __restrict__ W,
    const float* __restrict__ bias, void* __restrict__ out,
    int M, int N, int K) {
  constexpr int LDK = 40;                      // 32 + pad
  __shared__ __bf16 ldsA[128 * LDK];
  __shared__ __bf16 ldsB[128 * LDK];

  const int tid  = threadIdx.x;
  const int lane = tid & 31;
  const int wave = tid >> 5;
  const int half = lane >> 4;
  const int l16  = lane & 15;
  const int wm   = wave >> 1;                  // 0..3  (M direction)
  const int wn   = wave & 1;                   // 0..1  (N direction)

  const int mBase = blockIdx.x * 128;
  const int nBase = blockIdx.y * 128;

  v8f acc[2][4];
#pragma unroll
  for (int mt = 0; mt < 2; ++mt)
#pragma unroll
    for (int nt = 0; nt < 4; ++nt) acc[mt][nt] = vzero8();

  const int sr = tid >> 1;                     // staging row 0..127
  const int sc = (tid & 1) * 16;               // staging col 0 / 16

  for (int kb = 0; kb < K; kb += 32) {
    {
      const AT* src = A + (size_t)(mBase + sr) * K + kb + sc;
      __bf16* dst = &ldsA[sr * LDK + sc];
#pragma unroll
      for (int i = 0; i < 16; ++i) dst[i] = to_bf16(src[i]);
      __builtin_prefetch(src + 32, 0, 3);      // next K slice -> global_prefetch
    }
    {
      const float* src = W + (size_t)(nBase + sr) * K + kb + sc;
      __bf16* dst = &ldsB[sr * LDK + sc];
#pragma unroll
      for (int i = 0; i < 16; ++i) dst[i] = to_bf16(src[i]);
      __builtin_prefetch(src + 32, 0, 3);
    }
    __syncthreads();

    v16bf afrag[2], bfrag[4];
#pragma unroll
    for (int mt = 0; mt < 2; ++mt)
      afrag[mt] = load_frag_a(&ldsA[(wm * 32 + mt * 16 + l16) * LDK], half);
#pragma unroll
    for (int nt = 0; nt < 4; ++nt)
      bfrag[nt] = load_frag_b(&ldsB[(wn * 64 + nt * 16 + l16) * LDK], half);
#pragma unroll
    for (int mt = 0; mt < 2; ++mt)
#pragma unroll
      for (int nt = 0; nt < 4; ++nt)
        acc[mt][nt] = wmma_bf16(afrag[mt], bfrag[nt], acc[mt][nt]);
    __syncthreads();
  }

#pragma unroll
  for (int mt = 0; mt < 2; ++mt)
#pragma unroll
    for (int nt = 0; nt < 4; ++nt) {
      const int col = nBase + wn * 64 + nt * 16 + l16;
      const float bv = bias[col];
#pragma unroll
      for (int r8 = 0; r8 < 8; ++r8) {
        const int row = mBase + wm * 32 + mt * 16 + r8 + 8 * half;
        const float v = acc[mt][nt][r8] + bv;
        if (OUTMODE == 0) {
          const int b = row >> 11, s = row & (S_ - 1);
          const int h = col >> 6,  dk = col & (DK_ - 1);
          ((__bf16*)out)[(((size_t)b * H_ + h) * S_ + s) * DK_ + dk] = to_bf16(v);
        } else {
          ((float*)out)[(size_t)row * N + col] = v;
        }
      }
    }
}

// ---------------------------------------------------------------------------
// Flash attention: one block per (b*h, 128 q rows); 8 waves x 16 q rows.
// K tile staged by the Tensor Data Mover; V staged transposed by VALU/DS.
// ---------------------------------------------------------------------------
__global__ __launch_bounds__(256) void flash_attn(
    const __bf16* __restrict__ Qh,   // [B*H, S, DK] bf16
    const __bf16* __restrict__ Kh,
    const __bf16* __restrict__ Vh,
    const int* __restrict__ mask,    // [B, S, S]
    __bf16* __restrict__ attn) {     // [B*S, D] bf16 (heads concat)
  constexpr int KB  = 64;            // key tile
  constexpr int LDA = 72;            // 64 + pad (matches TDM pad descriptor)
  __shared__ __bf16 ldsK[KB * LDA];        // [key][d]
  __shared__ __bf16 ldsVT[DK_ * LDA];      // [d][key]
  __shared__ __bf16 ldsP[8 * 16 * LDA];    // per-wave P tile [16][64+pad]

  const int tid  = threadIdx.x;
  const int lane = tid & 31;
  const int wave = tid >> 5;
  const int half = lane >> 4;
  const int l16  = lane & 15;

  const int bh = blockIdx.y;         // b*H + h
  const int b  = bh >> 4;
  const int h  = bh & (H_ - 1);
  const int q0 = blockIdx.x * 128;
  const int qrow = q0 + wave * 16;   // first of the wave's 16 q rows

  // Q fragments (16 x 64 -> two 16x32 chunks), gathered straight from global.
  const __bf16* qbase = Qh + ((size_t)bh * S_ + qrow + l16) * DK_;
  v16bf qf[2];
#pragma unroll
  for (int kc = 0; kc < 2; ++kc) qf[kc] = load_frag_a(qbase + kc * 32, half);

  v8f o[4];
#pragma unroll
  for (int dt = 0; dt < 4; ++dt) o[dt] = vzero8();
  float mrun[8], lrun[8];
#pragma unroll
  for (int r8 = 0; r8 < 8; ++r8) { mrun[r8] = -1e30f; lrun[r8] = 0.f; }

  const int sr = tid >> 2;           // staging key row 0..63
  const int sc = (tid & 3) * 16;     // staging col 0/16/32/48
  const size_t kvbase = (size_t)bh * S_ * DK_;
#ifdef HAVE_TDM
  const unsigned ldsK_addr = (unsigned)(uintptr_t)&ldsK[0];
#endif

  for (int kb0 = 0; kb0 < S_; kb0 += KB) {
#ifdef HAVE_TDM
    if (wave == 0) {   // one TDM issue per block; EXEC is ignored by TDM
      tdm_load_2d_bf16(ldsK_addr, Kh + kvbase + (size_t)kb0 * DK_,
                       /*tensor_d0=*/DK_, /*tensor_d1=*/S_ - kb0,
                       /*tile_d0=*/DK_, /*tile_d1=*/KB, /*stride0=*/DK_);
    }
#else
    {
      const __bf16* ksrc = Kh + kvbase + (size_t)(kb0 + sr) * DK_ + sc;
      __bf16* kdst = &ldsK[sr * LDA + sc];
#pragma unroll
      for (int i = 0; i < 16; ++i) kdst[i] = ksrc[i];
    }
#endif
    {  // stage V transposed: [d][key]
      const __bf16* vsrc = Vh + kvbase + (size_t)(kb0 + sr) * DK_ + sc;
#pragma unroll
      for (int i = 0; i < 16; ++i) ldsVT[(sc + i) * LDA + sr] = vsrc[i];
    }
#ifdef HAVE_TDM
    if (wave == 0) __builtin_amdgcn_s_wait_tensorcnt(0);
#endif
    __syncthreads();

    // scores: 16 x 64 tile = 4 WMMA n-tiles, K = 64 (2 chunks)
    v8f sct[4];
#pragma unroll
    for (int nt = 0; nt < 4; ++nt) {
      v8f s = vzero8();
#pragma unroll
      for (int kc = 0; kc < 2; ++kc) {
        v16bf kf = load_frag_b(&ldsK[(nt * 16 + l16) * LDA + kc * 32], half);
        s = wmma_bf16(qf[kc], kf, s);
      }
      sct[nt] = s;
    }

    // scale + mask
    const float scale = 0.125f;      // 1/sqrt(64)
#pragma unroll
    for (int nt = 0; nt < 4; ++nt) {
      const int col = kb0 + nt * 16 + l16;
#pragma unroll
      for (int r8 = 0; r8 < 8; ++r8) {
        const int row = qrow + r8 + 8 * half;
        const float x = sct[nt][r8] * scale;
        const int mv = mask[(size_t)b * S_ * S_ + (size_t)row * S_ + col];
        sct[nt][r8] = (mv == 0) ? -1e9f : x;
      }
    }

    // online softmax update (row reductions live in 16-lane halves)
#pragma unroll
    for (int r8 = 0; r8 < 8; ++r8) {
      float mx = sct[0][r8];
#pragma unroll
      for (int nt = 1; nt < 4; ++nt) mx = fmaxf(mx, sct[nt][r8]);
#pragma unroll
      for (int off = 1; off < 16; off <<= 1)
        mx = fmaxf(mx, __shfl_xor(mx, off, 32));
      const float mnew = fmaxf(mrun[r8], mx);
      const float corr = __expf(mrun[r8] - mnew);
      mrun[r8] = mnew;
      lrun[r8] *= corr;
#pragma unroll
      for (int dt = 0; dt < 4; ++dt) o[dt][r8] *= corr;
    }

    // exp, write P tile to per-wave LDS, accumulate row sums
    __bf16* pbase = &ldsP[wave * 16 * LDA];
#pragma unroll
    for (int nt = 0; nt < 4; ++nt)
#pragma unroll
      for (int r8 = 0; r8 < 8; ++r8) {
        const float p = __expf(sct[nt][r8] - mrun[r8]);
        sct[nt][r8] = p;
        pbase[(r8 + 8 * half) * LDA + nt * 16 + l16] = to_bf16(p);
      }
#pragma unroll
    for (int r8 = 0; r8 < 8; ++r8) {
      float sm = 0.f;
#pragma unroll
      for (int nt = 0; nt < 4; ++nt) sm += sct[nt][r8];
#pragma unroll
      for (int off = 1; off < 16; off <<= 1)
        sm += __shfl_xor(sm, off, 32);
      lrun[r8] += sm;
    }

    // O += P (16x64) x V (64x64): A from ldsP, B from ldsVT
#pragma unroll
    for (int kc = 0; kc < 2; ++kc) {
      v16bf pf = load_frag_a(&pbase[l16 * LDA + kc * 32], half);
#pragma unroll
      for (int dt = 0; dt < 4; ++dt) {
        v16bf vf = load_frag_b(&ldsVT[(dt * 16 + l16) * LDA + kc * 32], half);
        o[dt] = wmma_bf16(pf, vf, o[dt]);
      }
    }
    __syncthreads();
  }

  // normalize and store concat-head output [B*S, D] as bf16
#pragma unroll
  for (int r8 = 0; r8 < 8; ++r8) {
    const float inv = 1.f / lrun[r8];
    const int srow = qrow + r8 + 8 * half;
#pragma unroll
    for (int dt = 0; dt < 4; ++dt) {
      const float v = o[dt][r8] * inv;
      attn[((size_t)b * S_ + srow) * D_ + h * DK_ + dt * 16 + l16] = to_bf16(v);
    }
  }
}

// ---------------------------------------------------------------------------
// Launcher. Inputs: q,k,v,mask,Wq,bq,Wk,bk,Wv,bv,Wo,bo.
// Workspace: Qh | Kh | Vh | attn, each B*S*D bf16 = 8 MiB -> 32 MiB total.
// ---------------------------------------------------------------------------
extern "C" void kernel_launch(void* const* d_in, const int* in_sizes, int n_in,
                              void* d_out, int out_size, void* d_ws, size_t ws_size,
                              hipStream_t stream) {
  const float* q    = (const float*)d_in[0];
  const float* k    = (const float*)d_in[1];
  const float* v    = (const float*)d_in[2];
  const int*   mask = (const int*)d_in[3];
  const float* Wq   = (const float*)d_in[4];
  const float* bq   = (const float*)d_in[5];
  const float* Wk   = (const float*)d_in[6];
  const float* bk   = (const float*)d_in[7];
  const float* Wv   = (const float*)d_in[8];
  const float* bv   = (const float*)d_in[9];
  const float* Wo   = (const float*)d_in[10];
  const float* bo   = (const float*)d_in[11];

  const size_t elems = (size_t)MROWS_ * D_;       // 4,194,304
  __bf16* qh   = (__bf16*)d_ws;
  __bf16* kh   = qh + elems;
  __bf16* vh   = kh + elems;
  __bf16* attn = vh + elems;

  dim3 gG(MROWS_ / 128, D_ / 128);                // 32 x 8
  gemm_bf16_wmma<float, 0><<<gG, 256, 0, stream>>>(q, Wq, bq, qh, MROWS_, D_, D_);
  gemm_bf16_wmma<float, 0><<<gG, 256, 0, stream>>>(k, Wk, bk, kh, MROWS_, D_, D_);
  gemm_bf16_wmma<float, 0><<<gG, 256, 0, stream>>>(v, Wv, bv, vh, MROWS_, D_, D_);

  dim3 gA(S_ / 128, B_ * H_);                     // 16 x 32
  flash_attn<<<gA, 256, 0, stream>>>(qh, kh, vh, mask, attn);

  gemm_bf16_wmma<__bf16, 1><<<gG, 256, 0, stream>>>(
      attn, Wo, bo, d_out, MROWS_, D_, D_);
}
